// APUNet_33079838114069
// MI455X (gfx1250) — matrix operand order repaired and checked by reading, not compiled, not run on hardware
//
#include <hip/hip_runtime.h>
#include <stdint.h>

// ---------------------------------------------------------------------------
// out[n,c,h,w] = x[n,c,h,w] + einsum('oi,nikj->nokj', G, unfold(x))
//   B=16, C=32, H=W=128, K=8, P=256  ->  GEMM M=8192, K=8192, N=1024
// Plan:
//   k1: G fp32 -> bf16 (row-major M x K)            [ws +0,    128 MB]
//   k2: pack x -> Xc bf16, column-major (N x K)     [ws +128MB, 16 MB]
//   k3: WMMA bf16 GEMM (f32 accum), double-buffered async global->LDS
//       staging (ASYNCcnt), padded LDS (bank-conflict free), fused
//       residual + inverse patch permute epilogue.
// ---------------------------------------------------------------------------

typedef __attribute__((ext_vector_type(16))) __bf16 v16bf;
typedef __attribute__((ext_vector_type(8)))  __bf16 v8bf;
typedef __attribute__((ext_vector_type(4)))  __bf16 v4bf;
typedef __attribute__((ext_vector_type(8)))  float  v8f;
typedef __attribute__((ext_vector_type(4)))  float  v4f;
typedef __attribute__((ext_vector_type(4)))  int    v4i;

#define MTOT 8192
#define NTOT 1024
#define KTOT 8192
#define BM   128    // block M tile
#define BN   256    // block N tile
#define BK   32     // K step (one 16x16x32 wmma deep)
#define LDA  40     // padded LDS row stride (bf16): 80B rows -> conflict-free
#define LDB  40

// ---- CDNA5 async global->LDS path (guarded; falls back to sync staging) ----
#if __has_builtin(__builtin_amdgcn_global_load_async_to_lds_b128)
#define HAVE_ASYNC_LDS 1
#endif

// builtin signature (from hipcc diagnostic): param0 = int4 * in AS1 (global),
// param1 = LDS side; payload typed as int4 for the b128 form.
typedef __attribute__((address_space(1))) v4i g_v4i;
typedef __attribute__((address_space(3))) v4i l_v4i;

static __device__ __forceinline__ void cp16_g2l(const __bf16* g, __bf16* l) {
#ifdef HAVE_ASYNC_LDS
  // 16B per lane, mem -> LDS, ASYNCcnt-tracked. Generic->AS cast via inttoptr:
  // low 32 bits of a flat LDS pointer are the LDS byte offset (ISA 10.2).
  __builtin_amdgcn_global_load_async_to_lds_b128(
      (g_v4i*)(uintptr_t)g, (l_v4i*)(uintptr_t)l, 0, 0);
#else
  *(v8bf*)l = *(const v8bf*)g;
#endif
}

static __device__ __forceinline__ void wait_staged() {
#ifdef HAVE_ASYNC_LDS
#if __has_builtin(__builtin_amdgcn_s_wait_asynccnt)
  __builtin_amdgcn_s_wait_asynccnt(0);
#else
  asm volatile("s_wait_asynccnt 0x0" ::: "memory");
#endif
#endif
}

static __device__ __forceinline__ __bf16 f32_to_bf16_rne(float f) {
  uint32_t u = __builtin_bit_cast(uint32_t, f);
  u += 0x7FFFu + ((u >> 16) & 1u);          // round to nearest even
  uint16_t h = (uint16_t)(u >> 16);
  return __builtin_bit_cast(__bf16, h);
}

// ---- k1: convert g_weight (fp32, 8192x8192 row-major) to bf16 --------------
__global__ void apunet_cvt_g(const float* __restrict__ g,
                             __bf16* __restrict__ gbf) {
  size_t t = (size_t)blockIdx.x * blockDim.x + threadIdx.x;   // 16M threads
  v4f v = *(const v4f*)(g + t * 4);
  v4bf o;
  o[0] = f32_to_bf16_rne(v[0]);
  o[1] = f32_to_bf16_rne(v[1]);
  o[2] = f32_to_bf16_rne(v[2]);
  o[3] = f32_to_bf16_rne(v[3]);
  *(v4bf*)(gbf + t * 4) = o;
}

// ---- k2: pack x into Xc[col][i] bf16 (col = n*64+k*8+j, i = c*256+hp*16+wp)
__global__ void apunet_pack_x(const float* __restrict__ x,
                              __bf16* __restrict__ xc) {
  size_t t = (size_t)blockIdx.x * blockDim.x + threadIdx.x;   // 8M threads
  int col = (int)(t >> 13);
  int i   = (int)(t & 8191);
  int n  = col >> 6, s  = col & 63;
  int kk = s >> 3,   jj = s & 7;
  int c  = i >> 8,   pp = i & 255;
  int hp = pp >> 4,  wp = pp & 15;
  size_t a = (((size_t)(n * 32 + c) * 128) + (size_t)(hp * 8 + kk)) * 128
           + (size_t)(wp * 8 + jj);
  xc[t] = f32_to_bf16_rne(x[a]);
}

// ---- k3: bf16 WMMA GEMM, 128x256 block tile, 8 wave32s, pipelined ----------
__global__ void __launch_bounds__(256)
apunet_wmma_gemm(const __bf16* __restrict__ gbf,   // [8192][8192]
                 const __bf16* __restrict__ xc,    // [1024][8192]  (B^T)
                 const float*  __restrict__ xres,  // residual (NCHW fp32)
                 float* __restrict__ out) {
  __shared__ __attribute__((aligned(64))) __bf16 As[2][BM * LDA];  // 2 x 10 KB
  __shared__ __attribute__((aligned(64))) __bf16 Bs[2][BN * LDB];  // 2 x 20 KB

  const int mb = blockIdx.x & 63;        // 64 M blocks
  const int nb = blockIdx.x >> 6;        // 4  N blocks
  const int m0 = mb * BM;
  const int n0 = nb * BN;

  const int tid  = threadIdx.x;
  const int wave = tid >> 5;             // 8 wave32s per WG
  const int lid  = tid & 31;
  const int wm0  = (wave & 3) * 32;      // wave M offset inside block
  const int wn0  = (wave >> 2) * 128;    // wave N offset inside block
  const int lrow = lid & 15;             // M/N index inside 16-wide tile
  const int kg   = lid >> 4;             // K half-group selector

  // stage one K-slab (A: 512, B: 1024 x 16B chunks) into LDS buffer `b`
  auto stage = [&](int k0, int b) {
#pragma unroll
    for (int r = 0; r < 2; ++r) {
      int idx = tid + r * 256;
      int row = idx >> 2, ch = idx & 3;
      cp16_g2l(&gbf[(size_t)(m0 + row) * KTOT + k0 + ch * 8],
               &As[b][row * LDA + ch * 8]);
    }
#pragma unroll
    for (int r = 0; r < 4; ++r) {
      int idx = tid + r * 256;
      int col = idx >> 2, ch = idx & 3;
      cp16_g2l(&xc[(size_t)(n0 + col) * KTOT + k0 + ch * 8],
               &Bs[b][col * LDB + ch * 8]);
    }
  };

  v8f acc[2][8];
#pragma unroll
  for (int tm = 0; tm < 2; ++tm)
#pragma unroll
    for (int tn = 0; tn < 8; ++tn)
      acc[tm][tn] = (v8f){0.f, 0.f, 0.f, 0.f, 0.f, 0.f, 0.f, 0.f};

  stage(0, 0);
  wait_staged();
  __syncthreads();

  int buf = 0;
  for (int k0 = 0; k0 < KTOT; k0 += BK, buf ^= 1) {
    // kick off next slab's async copies before touching this one
    if (k0 + BK < KTOT) stage(k0 + BK, buf ^ 1);
    // warm L2 two slabs ahead (global_prefetch_b8)
    if (k0 + 2 * BK < KTOT) {
      __builtin_prefetch(&gbf[(size_t)(m0 + (tid >> 1)) * KTOT + k0 + 2 * BK], 0, 3);
      __builtin_prefetch(&xc[(size_t)(n0 + tid) * KTOT + k0 + 2 * BK], 0, 3);
    }

    const __bf16* sA = As[buf];
    const __bf16* sB = Bs[buf];

    // A fragments: lane = row M=lrow; two 8-elt runs at K = kg*8 and kg*8+16
    v16bf a[2];
#pragma unroll
    for (int tm = 0; tm < 2; ++tm) {
      const __bf16* ap = &sA[(wm0 + tm * 16 + lrow) * LDA + kg * 8];
      v8bf lo = *(const v8bf*)ap;
      v8bf hi = *(const v8bf*)(ap + 16);
      a[tm] = __builtin_shufflevector(lo, hi, 0, 1, 2, 3, 4, 5, 6, 7,
                                      8, 9, 10, 11, 12, 13, 14, 15);
    }
    // B fragments: lane = col N=lrow; two 8-elt runs at K = kg*8 and kg*8+16
#pragma unroll
    for (int tn = 0; tn < 8; ++tn) {
      const __bf16* bp = &sB[(wn0 + tn * 16 + lrow) * LDB + kg * 8];
      v8bf blo = *(const v8bf*)bp;
      v8bf bhi = *(const v8bf*)(bp + 16);
      v16bf b = __builtin_shufflevector(blo, bhi, 0, 1, 2, 3, 4, 5, 6, 7,
                                        8, 9, 10, 11, 12, 13, 14, 15);
#pragma unroll
      for (int tm = 0; tm < 2; ++tm) {
        acc[tm][tn] = __builtin_amdgcn_wmma_f32_16x16x32_bf16(
            false, a[tm], false, b, (short)0, acc[tm][tn], false, false);
      }
    }

    wait_staged();      // next slab's async copies landed in LDS
    __syncthreads();    // everyone done reading `buf` before it is re-staged
  }

  // ---- epilogue: D element r -> M = r + 8*kg, N = lrow; inverse permute +
  //      fused residual add:  out[n,co,hp*8+k,wp*8+j] = D[o,col] + x[...]
#pragma unroll
  for (int tm = 0; tm < 2; ++tm) {
#pragma unroll
    for (int tn = 0; tn < 8; ++tn) {
      v8f d = acc[tm][tn];
      int gcol = n0 + wn0 + tn * 16 + lrow;
      int n  = gcol >> 6, s  = gcol & 63;
      int kk = s >> 3,    jj = s & 7;
#pragma unroll
      for (int r = 0; r < 8; ++r) {
        int o  = m0 + wm0 + tm * 16 + 8 * kg + r;
        int co = o >> 8, po = o & 255;
        int hp = po >> 4, wp = po & 15;
        size_t addr = (((size_t)(n * 32 + co) * 128) + (size_t)(hp * 8 + kk)) * 128
                    + (size_t)(wp * 8 + jj);
        out[addr] = d[r] + xres[addr];
      }
    }
  }
}

extern "C" void kernel_launch(void* const* d_in, const int* in_sizes, int n_in,
                              void* d_out, int out_size, void* d_ws, size_t ws_size,
                              hipStream_t stream) {
  const float* x = (const float*)d_in[0];   // [16,32,128,128] fp32
  const float* g = (const float*)d_in[1];   // [8192,8192] fp32
  float* out = (float*)d_out;               // [16,32,128,128] fp32

  // workspace: bf16 G (128 MB) then bf16 Xc (16 MB) -> needs 144 MB of d_ws
  __bf16* gbf = (__bf16*)d_ws;
  __bf16* xc  = (__bf16*)((char*)d_ws + (size_t)MTOT * KTOT * sizeof(__bf16));

  apunet_cvt_g<<<dim3((MTOT * (size_t)KTOT) / 4 / 256), dim3(256), 0, stream>>>(g, gbf);
  apunet_pack_x<<<dim3((NTOT * (size_t)KTOT) / 256), dim3(256), 0, stream>>>(x, xc);
  apunet_wmma_gemm<<<dim3(64 * 4), dim3(256), 0, stream>>>(gbf, xc, x, out);
}
